// DecoderBlock_50861002719352
// MI455X (gfx1250) — compile-verified, matrix operand order checked by reading, not run on hardware
//
#include <hip/hip_runtime.h>

// ---------------------------------------------------------------------------
// Types for CDNA5 WMMA (gfx1250, wave32)
// ---------------------------------------------------------------------------
typedef __bf16 bf16_t;
typedef __attribute__((ext_vector_type(16))) __bf16 v16bf;
typedef __attribute__((ext_vector_type(8)))  float  v8f;

union Frag16 {
    v16bf v;
    uint4 q[2];
};

__device__ __forceinline__ v8f wmma_bf16(const Frag16& a, const Frag16& b, v8f c) {
    // v_wmma_f32_16x16x32_bf16 : D = A(16x32) * B(32x16) + C(16x16 f32)
    return __builtin_amdgcn_wmma_f32_16x16x32_bf16(
        false, a.v, false, b.v, (short)0, c, false, false);
}

// ---------------------------------------------------------------------------
// Optional gfx1250 async global->LDS path (compile-safe feature probe)
// Builtin signature (from hipcc diagnostic): (int4 AS1*, int4 AS3*, imm, imm)
// ---------------------------------------------------------------------------
#if defined(__HIP_DEVICE_COMPILE__)
#if __has_builtin(__builtin_amdgcn_global_load_async_to_lds_b128)
#define HAVE_ASYNC_LDS 1
#endif
#endif
#ifndef HAVE_ASYNC_LDS
#define HAVE_ASYNC_LDS 0
#endif

#if HAVE_ASYNC_LDS
typedef int v4i_vs __attribute__((vector_size(16)));
typedef __attribute__((address_space(1))) v4i_vs* as1_v4i;
typedef __attribute__((address_space(3))) v4i_vs* as3_v4i;
__device__ __forceinline__ void async_copy16(void* lds_dst, const void* glb_src) {
    __builtin_amdgcn_global_load_async_to_lds_b128(
        (as1_v4i)glb_src, (as3_v4i)lds_dst, 0, 0);
}
__device__ __forceinline__ void wait_async0() {
#if __has_builtin(__builtin_amdgcn_s_wait_asynccnt)
    __builtin_amdgcn_s_wait_asynccnt(0);
#else
    asm volatile("s_wait_asynccnt 0x0" ::: "memory");
#endif
}
#endif

// ---------------------------------------------------------------------------
// Problem constants
// ---------------------------------------------------------------------------
#define NB      2
#define LSEQ    2048
#define DMODEL  2048
#define NHEADS  16
#define DHEAD   128
#define FFDIM   8192
#define NTOK    (NB * LSEQ)          // 4096 rows

// ---------------------------------------------------------------------------
// Weight transpose + f32 -> bf16 convert:  in[K][N] f32  ->  out[N][K] bf16
// ---------------------------------------------------------------------------
__global__ void tconv_kernel(const float* __restrict__ in, bf16_t* __restrict__ out,
                             int K, int N) {
    __shared__ float tile[32][33];
    const int kb = blockIdx.y * 32, nb = blockIdx.x * 32;
    const int tx = threadIdx.x & 31, ty = threadIdx.x >> 5;   // 32 x 8
    #pragma unroll
    for (int i = 0; i < 32; i += 8)
        tile[ty + i][tx] = in[(size_t)(kb + ty + i) * N + nb + tx];
    __syncthreads();
    #pragma unroll
    for (int i = 0; i < 32; i += 8)
        out[(size_t)(nb + ty + i) * K + kb + tx] = (bf16_t)tile[tx][ty + i];
}

// ---------------------------------------------------------------------------
// RMSNorm: row-wise, f32 in -> bf16 out
// ---------------------------------------------------------------------------
__global__ void rmsnorm_kernel(const float* __restrict__ x, const float* __restrict__ w,
                               bf16_t* __restrict__ out, int D) {
    const int row = blockIdx.x;
    const float* xr = x + (size_t)row * D;
    float ss = 0.f;
    for (int i = threadIdx.x; i < D; i += blockDim.x) {
        float v = xr[i];
        ss += v * v;
    }
    #pragma unroll
    for (int off = 16; off > 0; off >>= 1) ss += __shfl_xor(ss, off, 32);
    __shared__ float red[8];
    if ((threadIdx.x & 31) == 0) red[threadIdx.x >> 5] = ss;
    __syncthreads();
    float tot = red[0];
    #pragma unroll
    for (int i = 1; i < 8; ++i) tot += red[i];
    const float inv = rsqrtf(tot / (float)D + 1.1920929e-07f);
    for (int i = threadIdx.x; i < D; i += blockDim.x)
        out[(size_t)row * D + i] = (bf16_t)(xr[i] * inv * w[i]);
}

// ---------------------------------------------------------------------------
// RoPE (in-place on bf16 q,k; q pre-scaled by dh^-0.5) + V transpose per head
//   q,k,v : [B, L, H, DH] bf16      vT : [B, H, DH, L] bf16
// ---------------------------------------------------------------------------
__global__ void rope_kernel(bf16_t* __restrict__ q, bf16_t* __restrict__ k,
                            const bf16_t* __restrict__ v, bf16_t* __restrict__ vT,
                            const float* __restrict__ cost, const float* __restrict__ sint) {
    const int HALF = DHEAD / 2;
    size_t idx = (size_t)blockIdx.x * blockDim.x + threadIdx.x;
    const int p = (int)(idx & (HALF - 1));          // 0..63
    size_t t = idx >> 6;
    const int h = (int)(t & (NHEADS - 1));  t >>= 4;
    const int l = (int)(t & (LSEQ - 1));
    const int b = (int)(t >> 11);
    const int d1 = p, d2 = p + HALF;
    const size_t base = ((size_t)(b * LSEQ + l) * NHEADS + h) * DHEAD;

    const float c1 = cost[l * DHEAD + d1], s1 = sint[l * DHEAD + d1];
    const float c2 = cost[l * DHEAD + d2], s2 = sint[l * DHEAD + d2];
    const float qs = 0.08838834764831845f;          // 128^-0.5 folded into Q

    float q1 = (float)q[base + d1], q2 = (float)q[base + d2];
    q[base + d1] = (bf16_t)((q1 * c1 - q2 * s1) * qs);
    q[base + d2] = (bf16_t)((q2 * c2 + q1 * s2) * qs);

    float k1 = (float)k[base + d1], k2 = (float)k[base + d2];
    k[base + d1] = (bf16_t)(k1 * c1 - k2 * s1);
    k[base + d2] = (bf16_t)(k2 * c2 + k1 * s2);

    const size_t vtb = ((size_t)b * NHEADS + h) * DHEAD;
    vT[(vtb + d1) * LSEQ + l] = v[base + d1];
    vT[(vtb + d2) * LSEQ + l] = v[base + d2];
}

// ---------------------------------------------------------------------------
// Tiled bf16 WMMA GEMM:  C[M,N] = A[M,K] @ Bt[N,K]^T  (+ epilogue)
//   EPI 0: store bf16          EPI 1: f32 store with residual add
//   EPI 2: SiLU -> bf16
// Block: 256 thr = 8 waves (4x2), tile 256x128, BK=32, wave tile 64x64.
// Double-buffered LDS; async global->LDS when the toolchain exposes it.
// ---------------------------------------------------------------------------
#define BM 256
#define BN 128
#define BKK 32
#define LDT (BKK + 8)   // 40 elements -> 80B row stride (16B aligned)

template <int EPI>
__global__ __launch_bounds__(256) void gemm_bf16_kernel(
    const bf16_t* __restrict__ A, const bf16_t* __restrict__ Bt,
    void* __restrict__ C, const float* __restrict__ resid,
    int M, int N, int K) {
    __shared__ bf16_t As[2][BM][LDT];   // 40960 B
    __shared__ bf16_t Bs[2][BN][LDT];   // 20480 B

    const int tid  = threadIdx.x;
    const int lane = tid & 31;
    const int wave = tid >> 5;
    const int wm   = wave >> 1;          // 0..3  (64-row group)
    const int wn   = wave & 1;           // 0..1  (64-col group)
    const int rowBase = blockIdx.y * BM;
    const int colBase = blockIdx.x * BN;
    const int lmr = lane & 15;
    const int lhk = (lane >> 4) * 8;

    // staging coordinates: 8-element (16B) chunks, 4 chunks per 32-wide row
    int ra[4], ca[4], rb[2], cb[2];
    #pragma unroll
    for (int i = 0; i < 4; ++i) {
        const int linear = tid + i * 256;
        ra[i] = linear >> 2;
        ca[i] = (linear & 3) * 8;
    }
    #pragma unroll
    for (int i = 0; i < 2; ++i) {
        const int linear = tid + i * 256;
        rb[i] = linear >> 2;
        cb[i] = (linear & 3) * 8;
    }

    v8f acc[4][4];
    #pragma unroll
    for (int mt = 0; mt < 4; ++mt)
        #pragma unroll
        for (int nt = 0; nt < 4; ++nt) acc[mt][nt] = 0.f;

    const int nChunks = K / BKK;

#if HAVE_ASYNC_LDS
    // ---- async pipeline prologue: stage chunk 0 ----
    #pragma unroll
    for (int i = 0; i < 4; ++i)
        async_copy16(&As[0][ra[i]][ca[i]], &A[(size_t)(rowBase + ra[i]) * K + ca[i]]);
    #pragma unroll
    for (int i = 0; i < 2; ++i)
        async_copy16(&Bs[0][rb[i]][cb[i]], &Bt[(size_t)(colBase + rb[i]) * K + cb[i]]);
    wait_async0();
    __syncthreads();
#else
    // ---- register-staged prologue: stage chunk 0 ----
    uint4 pa[4], pb[2];
    #pragma unroll
    for (int i = 0; i < 4; ++i)
        pa[i] = *(const uint4*)&A[(size_t)(rowBase + ra[i]) * K + ca[i]];
    #pragma unroll
    for (int i = 0; i < 2; ++i)
        pb[i] = *(const uint4*)&Bt[(size_t)(colBase + rb[i]) * K + cb[i]];
    #pragma unroll
    for (int i = 0; i < 4; ++i) *(uint4*)&As[0][ra[i]][ca[i]] = pa[i];
    #pragma unroll
    for (int i = 0; i < 2; ++i) *(uint4*)&Bs[0][rb[i]][cb[i]] = pb[i];
    __syncthreads();
#endif

    int cur = 0;
    for (int ch = 0; ch < nChunks; ++ch) {
        const bool more  = (ch + 1) < nChunks;
        const int  knext = (ch + 1) * BKK;

        if (more) {
#if HAVE_ASYNC_LDS
            // issue async loads for the next chunk into the other buffer
            #pragma unroll
            for (int i = 0; i < 4; ++i)
                async_copy16(&As[cur ^ 1][ra[i]][ca[i]],
                             &A[(size_t)(rowBase + ra[i]) * K + knext + ca[i]]);
            #pragma unroll
            for (int i = 0; i < 2; ++i)
                async_copy16(&Bs[cur ^ 1][rb[i]][cb[i]],
                             &Bt[(size_t)(colBase + rb[i]) * K + knext + cb[i]]);
#else
            #pragma unroll
            for (int i = 0; i < 4; ++i)
                pa[i] = *(const uint4*)&A[(size_t)(rowBase + ra[i]) * K + knext + ca[i]];
            #pragma unroll
            for (int i = 0; i < 2; ++i)
                pb[i] = *(const uint4*)&Bt[(size_t)(colBase + rb[i]) * K + knext + cb[i]];
#endif
        }

        // ---- compute on current buffer: 16 WMMAs / wave / chunk ----
        Frag16 af[4], bfr[4];
        #pragma unroll
        for (int mt = 0; mt < 4; ++mt) {
            const int r = wm * 64 + mt * 16 + lmr;
            af[mt].q[0] = *(const uint4*)&As[cur][r][lhk];
            af[mt].q[1] = *(const uint4*)&As[cur][r][lhk + 16];
        }
        #pragma unroll
        for (int nt = 0; nt < 4; ++nt) {
            const int r = wn * 64 + nt * 16 + lmr;
            bfr[nt].q[0] = *(const uint4*)&Bs[cur][r][lhk];
            bfr[nt].q[1] = *(const uint4*)&Bs[cur][r][lhk + 16];
        }
        #pragma unroll
        for (int mt = 0; mt < 4; ++mt)
            #pragma unroll
            for (int nt = 0; nt < 4; ++nt)
                acc[mt][nt] = wmma_bf16(af[mt], bfr[nt], acc[mt][nt]);

        if (more) {
#if HAVE_ASYNC_LDS
            wait_async0();
#else
            #pragma unroll
            for (int i = 0; i < 4; ++i) *(uint4*)&As[cur ^ 1][ra[i]][ca[i]] = pa[i];
            #pragma unroll
            for (int i = 0; i < 2; ++i) *(uint4*)&Bs[cur ^ 1][rb[i]][cb[i]] = pb[i];
#endif
        }
        __syncthreads();
        cur ^= 1;
    }

    // ---- epilogue ----
    const int rowSub = (lane >> 4) * 8;
    const int colSub = lane & 15;
    #pragma unroll
    for (int mt = 0; mt < 4; ++mt)
        #pragma unroll
        for (int nt = 0; nt < 4; ++nt)
            #pragma unroll
            for (int r = 0; r < 8; ++r) {
                const int row = rowBase + wm * 64 + mt * 16 + rowSub + r;
                const int col = colBase + wn * 64 + nt * 16 + colSub;
                const size_t o = (size_t)row * N + col;
                const float v = acc[mt][nt][r];
                if constexpr (EPI == 0) {
                    ((bf16_t*)C)[o] = (bf16_t)v;
                } else if constexpr (EPI == 1) {
                    ((float*)C)[o] = v + resid[o];
                } else {
                    ((bf16_t*)C)[o] = (bf16_t)(v / (1.f + __expf(-v)));
                }
            }
}

// ---------------------------------------------------------------------------
// Flash attention (causal + key mask), bf16 WMMA, online softmax.
// Block: 128 thr = 4 waves; 64 queries/block; KV streamed in 64-key tiles.
//   q,k : [B,L,H,DH] bf16 (q pre-scaled)   vT : [B,H,DH,L] bf16
//   out : [B,L,H,DH] bf16
// ---------------------------------------------------------------------------
__global__ __launch_bounds__(128) void attn_kernel(
    const bf16_t* __restrict__ q, const bf16_t* __restrict__ k,
    const bf16_t* __restrict__ vT, const int* __restrict__ mask,
    bf16_t* __restrict__ out) {
    const int qblk = blockIdx.x;
    const int h    = blockIdx.y;
    const int b    = blockIdx.z;
    const int tid  = threadIdx.x;
    const int lane = tid & 31;
    const int wave = tid >> 5;           // 0..3, each owns 16 query rows
    const int qbase = qblk * 64;
    const int lmr = lane & 15;
    const int lhk = (lane >> 4) * 8;
    const int rowSub = (lane >> 4) * 8;
    const int colSub = lane & 15;

    __shared__ bf16_t Ks[64][DHEAD + 8];    // [key][dh]
    __shared__ bf16_t Vs[DHEAD][64 + 8];    // [dh][key]  (V^T tile)
    __shared__ bf16_t Ps[4][16][64 + 8];    // per-wave P

    // Q fragments resident in registers: 16 rows x 128 dh (4 k-steps)
    Frag16 qf[4];
    {
        const size_t qrow = ((size_t)(b * LSEQ + qbase + wave * 16 + lmr) * NHEADS + h) * DHEAD;
        #pragma unroll
        for (int kk = 0; kk < 4; ++kk) {
            qf[kk].q[0] = *(const uint4*)&q[qrow + kk * 32 + lhk];
            qf[kk].q[1] = *(const uint4*)&q[qrow + kk * 32 + lhk + 16];
        }
    }

    v8f o[8];
    #pragma unroll
    for (int nt = 0; nt < 8; ++nt) o[nt] = 0.f;
    float mrow[8], lrow[8];
    #pragma unroll
    for (int r = 0; r < 8; ++r) { mrow[r] = -1e30f; lrow[r] = 0.f; }

    for (int kv = 0; kv <= qbase; kv += 64) {
        // stage K tile: 64 keys x 128 dh
        #pragma unroll
        for (int i = 0; i < 8; ++i) {
            int linear = tid + i * 128;
            int r = linear >> 4;
            int c = (linear & 15) * 8;
            *(uint4*)&Ks[r][c] =
                *(const uint4*)&k[((size_t)(b * LSEQ + kv + r) * NHEADS + h) * DHEAD + c];
        }
        // stage V^T tile: 128 dh x 64 keys
        #pragma unroll
        for (int i = 0; i < 8; ++i) {
            int linear = tid + i * 128;
            int r = linear >> 3;
            int c = (linear & 7) * 8;
            *(uint4*)&Vs[r][c] =
                *(const uint4*)&vT[(((size_t)b * NHEADS + h) * DHEAD + r) * LSEQ + kv + c];
        }
        __syncthreads();

        // S = Q @ K^T  (16 q-rows x 64 keys per wave)
        v8f s[4];
        #pragma unroll
        for (int nt = 0; nt < 4; ++nt) s[nt] = 0.f;
        #pragma unroll
        for (int kk = 0; kk < 4; ++kk) {
            #pragma unroll
            for (int nt = 0; nt < 4; ++nt) {
                Frag16 bk;
                const int kr = nt * 16 + lmr;
                bk.q[0] = *(const uint4*)&Ks[kr][kk * 32 + lhk];
                bk.q[1] = *(const uint4*)&Ks[kr][kk * 32 + lhk + 16];
                s[nt] = wmma_bf16(qf[kk], bk, s[nt]);
            }
        }

        // causal + key mask, online softmax (row = qbase + wave*16 + rowSub + r)
        #pragma unroll
        for (int r = 0; r < 8; ++r) {
            const int qg = qbase + wave * 16 + rowSub + r;
            float rowmax = -1e30f;
            #pragma unroll
            for (int nt = 0; nt < 4; ++nt) {
                const int kg = kv + nt * 16 + colSub;
                float sv = s[nt][r];
                const bool ok = (kg <= qg) && (mask[b * LSEQ + kg] != 0);
                sv = ok ? sv : -1e30f;
                s[nt][r] = sv;
                rowmax = fmaxf(rowmax, sv);
            }
            #pragma unroll
            for (int off = 1; off < 16; off <<= 1)
                rowmax = fmaxf(rowmax, __shfl_xor(rowmax, off, 32));
            const float mnew = fmaxf(mrow[r], rowmax);
            const float scale = __expf(mrow[r] - mnew);
            float rsum = 0.f;
            #pragma unroll
            for (int nt = 0; nt < 4; ++nt) {
                const float p = __expf(s[nt][r] - mnew);
                s[nt][r] = p;
                rsum += p;
            }
            #pragma unroll
            for (int off = 1; off < 16; off <<= 1)
                rsum += __shfl_xor(rsum, off, 32);
            lrow[r] = lrow[r] * scale + rsum;
            mrow[r] = mnew;
            #pragma unroll
            for (int nt = 0; nt < 8; ++nt) o[nt][r] *= scale;
        }

        // P (C-layout f32) -> LDS bf16 -> A-fragment layout
        #pragma unroll
        for (int nt = 0; nt < 4; ++nt)
            #pragma unroll
            for (int r = 0; r < 8; ++r)
                Ps[wave][rowSub + r][nt * 16 + colSub] = (bf16_t)s[nt][r];

        // O += P @ V   (2 k-steps over 64 keys, 8 dh tiles)
        #pragma unroll
        for (int ks = 0; ks < 2; ++ks) {
            Frag16 pa;
            pa.q[0] = *(const uint4*)&Ps[wave][lmr][ks * 32 + lhk];
            pa.q[1] = *(const uint4*)&Ps[wave][lmr][ks * 32 + lhk + 16];
            #pragma unroll
            for (int nt = 0; nt < 8; ++nt) {
                Frag16 vb;
                const int vr = nt * 16 + lmr;
                vb.q[0] = *(const uint4*)&Vs[vr][ks * 32 + lhk];
                vb.q[1] = *(const uint4*)&Vs[vr][ks * 32 + lhk + 16];
                o[nt] = wmma_bf16(pa, vb, o[nt]);
            }
        }
        __syncthreads();
    }

    // normalize + store: out[b, qg, h, dh]
    #pragma unroll
    for (int nt = 0; nt < 8; ++nt)
        #pragma unroll
        for (int r = 0; r < 8; ++r) {
            const int qg = qbase + wave * 16 + rowSub + r;
            const int dh = nt * 16 + colSub;
            out[((size_t)(b * LSEQ + qg) * NHEADS + h) * DHEAD + dh] =
                (bf16_t)(o[nt][r] / lrow[r]);
        }
}

// ---------------------------------------------------------------------------
// Host orchestration
// ---------------------------------------------------------------------------
extern "C" void kernel_launch(void* const* d_in, const int* in_sizes, int n_in,
                              void* d_out, int out_size, void* d_ws, size_t ws_size,
                              hipStream_t stream) {
    const float* x      = (const float*)d_in[0];
    const float* ropec  = (const float*)d_in[1];
    const float* ropes  = (const float*)d_in[2];
    const int*   maskp  = (const int*)d_in[3];
    const float* wn1    = (const float*)d_in[4];
    const float* wn2    = (const float*)d_in[5];
    const float* wq     = (const float*)d_in[6];
    const float* wk     = (const float*)d_in[7];
    const float* wv     = (const float*)d_in[8];
    const float* wo     = (const float*)d_in[9];
    const float* wff1   = (const float*)d_in[10];
    const float* wff2   = (const float*)d_in[11];

    char* ws = (char*)d_ws;
    size_t off = 0;
    auto alloc = [&](size_t bytes) -> void* {
        void* p = ws + off;
        off = (off + bytes + 255) & ~(size_t)255;
        return p;
    };

    bf16_t* wqT  = (bf16_t*)alloc((size_t)DMODEL * DMODEL * 2);
    bf16_t* wkT  = (bf16_t*)alloc((size_t)DMODEL * DMODEL * 2);
    bf16_t* wvT  = (bf16_t*)alloc((size_t)DMODEL * DMODEL * 2);
    bf16_t* woT  = (bf16_t*)alloc((size_t)DMODEL * DMODEL * 2);
    bf16_t* wf1T = (bf16_t*)alloc((size_t)DMODEL * FFDIM * 2);
    bf16_t* wf2T = (bf16_t*)alloc((size_t)DMODEL * FFDIM * 2);
    bf16_t* xn   = (bf16_t*)alloc((size_t)NTOK * DMODEL * 2);
    bf16_t* qb   = (bf16_t*)alloc((size_t)NTOK * DMODEL * 2);
    bf16_t* kb   = (bf16_t*)alloc((size_t)NTOK * DMODEL * 2);
    bf16_t* vb   = (bf16_t*)alloc((size_t)NTOK * DMODEL * 2);
    bf16_t* vT   = (bf16_t*)alloc((size_t)NTOK * DMODEL * 2);
    bf16_t* attn = (bf16_t*)alloc((size_t)NTOK * DMODEL * 2);
    float*  x2   = (float*) alloc((size_t)NTOK * DMODEL * 4);
    bf16_t* xn2  = (bf16_t*)alloc((size_t)NTOK * DMODEL * 2);
    bf16_t* hb   = (bf16_t*)alloc((size_t)NTOK * FFDIM * 2);
    (void)ws_size; (void)in_sizes; (void)n_in; (void)out_size;

    // 1) weights: f32 [K,N] -> bf16 [N,K]
    tconv_kernel<<<dim3(DMODEL / 32, DMODEL / 32), 256, 0, stream>>>(wq, wqT, DMODEL, DMODEL);
    tconv_kernel<<<dim3(DMODEL / 32, DMODEL / 32), 256, 0, stream>>>(wk, wkT, DMODEL, DMODEL);
    tconv_kernel<<<dim3(DMODEL / 32, DMODEL / 32), 256, 0, stream>>>(wv, wvT, DMODEL, DMODEL);
    tconv_kernel<<<dim3(DMODEL / 32, DMODEL / 32), 256, 0, stream>>>(wo, woT, DMODEL, DMODEL);
    tconv_kernel<<<dim3(FFDIM / 32, DMODEL / 32), 256, 0, stream>>>(wff1, wf1T, DMODEL, FFDIM);
    tconv_kernel<<<dim3(DMODEL / 32, FFDIM / 32), 256, 0, stream>>>(wff2, wf2T, FFDIM, DMODEL);

    // 2) xn = rmsnorm(x)
    rmsnorm_kernel<<<NTOK, 256, 0, stream>>>(x, wn1, xn, DMODEL);

    // 3) q/k/v = xn @ W  (bf16 out)
    dim3 gD(DMODEL / BN, NTOK / BM);
    gemm_bf16_kernel<0><<<gD, 256, 0, stream>>>(xn, wqT, qb, nullptr, NTOK, DMODEL, DMODEL);
    gemm_bf16_kernel<0><<<gD, 256, 0, stream>>>(xn, wkT, kb, nullptr, NTOK, DMODEL, DMODEL);
    gemm_bf16_kernel<0><<<gD, 256, 0, stream>>>(xn, wvT, vb, nullptr, NTOK, DMODEL, DMODEL);

    // 4) RoPE (q,k in place) + V transpose
    {
        const size_t pairs = (size_t)NTOK * NHEADS * (DHEAD / 2);
        rope_kernel<<<(unsigned)(pairs / 256), 256, 0, stream>>>(qb, kb, vb, vT, ropec, ropes);
    }

    // 5) flash attention
    attn_kernel<<<dim3(LSEQ / 64, NHEADS, NB), 128, 0, stream>>>(qb, kb, vT, maskp, attn);

    // 6) x2 = x + attn @ wo
    gemm_bf16_kernel<1><<<gD, 256, 0, stream>>>(attn, woT, x2, x, NTOK, DMODEL, DMODEL);

    // 7) xn2 = rmsnorm(x2)
    rmsnorm_kernel<<<NTOK, 256, 0, stream>>>(x2, wn2, xn2, DMODEL);

    // 8) hb = silu(xn2 @ w_ff1)
    gemm_bf16_kernel<2><<<dim3(FFDIM / BN, NTOK / BM), 256, 0, stream>>>(
        xn2, wf1T, hb, nullptr, NTOK, FFDIM, DMODEL);

    // 9) out = x2 + hb @ w_ff2
    gemm_bf16_kernel<1><<<gD, 256, 0, stream>>>(hb, wf2T, (float*)d_out, x2, NTOK, DMODEL, FFDIM);
}
